// FSCM_16260746183043
// MI455X (gfx1250) — matrix operand order repaired and checked by reading, not compiled, not run on hardware
//
#include <hip/hip_runtime.h>
#include <hip/hip_bf16.h>

typedef __attribute__((ext_vector_type(16))) _Float16 v16h;
typedef __attribute__((ext_vector_type(8)))  float    v8f;

#define WAVE 32

// Fast transcendentals: v_exp_f32 / v_rcp_f32 (TRANS32, co-executes with VALU)
__device__ __forceinline__ float fast_sigmoid(float x) {
  return __builtin_amdgcn_rcpf(1.f + __expf(-x));
}

// ---------------------------------------------------------------------------
// Reductions (wave32)
// ---------------------------------------------------------------------------
__device__ __forceinline__ float wave_sum(float v) {
#pragma unroll
  for (int o = 16; o > 0; o >>= 1) v += __shfl_down(v, o, WAVE);
  return v;
}

__device__ __forceinline__ float block_sum(float v, float* red) {
  const int lane = threadIdx.x & (WAVE - 1);
  const int wid  = threadIdx.x >> 5;
  v = wave_sum(v);
  if (lane == 0) red[wid] = v;
  __syncthreads();
  if (wid == 0) {
    float t = (lane < (int)(blockDim.x >> 5)) ? red[lane] : 0.f;
    t = wave_sum(t);
    if (lane == 0) red[0] = t;
  }
  __syncthreads();
  const float r = red[0];
  __syncthreads();
  return r;
}

// ---------------------------------------------------------------------------
// SpatialGroupEnhance (groups=64). One block per (batch, group). float4 I/O.
// ---------------------------------------------------------------------------
__global__ __launch_bounds__(256) void sge_kernel(
    const float* __restrict__ x, const float* __restrict__ w,
    const float* __restrict__ b, float* __restrict__ out,
    int cpg, int HW)
{
  __shared__ float red[8];
  const int bg = blockIdx.x;
  const int g  = bg & 63;
  const float* xg = x   + (size_t)bg * cpg * HW;
  float*       og = out + (size_t)bg * cpg * HW;
  const int tid = threadIdx.x;
  const int HW4 = HW >> 2;

  float cmean[4];
  for (int c = 0; c < cpg; ++c) {
    const float4* xc = (const float4*)(xg + (size_t)c * HW);
    float s = 0.f;
    for (int i = tid; i < HW4; i += blockDim.x) {
      const float4 q = xc[i];
      s += q.x + q.y + q.z + q.w;
    }
    cmean[c] = block_sum(s, red) / (float)HW;
  }
  float s1 = 0.f, s2 = 0.f;
  for (int i = tid; i < HW4; i += blockDim.x) {
    float4 xn = {0.f, 0.f, 0.f, 0.f};
    for (int c = 0; c < cpg; ++c) {
      const float4 q = ((const float4*)(xg + (size_t)c * HW))[i];
      xn.x += q.x * cmean[c]; xn.y += q.y * cmean[c];
      xn.z += q.z * cmean[c]; xn.w += q.w * cmean[c];
    }
    s1 += xn.x + xn.y + xn.z + xn.w;
    s2 += xn.x*xn.x + xn.y*xn.y + xn.z*xn.z + xn.w*xn.w;
  }
  s1 = block_sum(s1, red);
  s2 = block_sum(s2, red);
  const float mean = s1 / (float)HW;
  const float var  = fmaxf((s2 - (float)HW * mean * mean) / (float)(HW - 1), 0.f);
  const float inv  = 1.f / (sqrtf(var) + 1e-5f);   // torch: t / (std + 1e-5)
  const float wg = w[g], bgv = b[g];
  for (int i = tid; i < HW4; i += blockDim.x) {
    float4 q[4];
    float4 xn = {0.f, 0.f, 0.f, 0.f};
    for (int c = 0; c < cpg; ++c) {
      q[c] = ((const float4*)(xg + (size_t)c * HW))[i];
      xn.x += q[c].x * cmean[c]; xn.y += q[c].y * cmean[c];
      xn.z += q[c].z * cmean[c]; xn.w += q[c].w * cmean[c];
    }
    float4 sg;
    sg.x = fast_sigmoid((xn.x - mean) * inv * wg + bgv);
    sg.y = fast_sigmoid((xn.y - mean) * inv * wg + bgv);
    sg.z = fast_sigmoid((xn.z - mean) * inv * wg + bgv);
    sg.w = fast_sigmoid((xn.w - mean) * inv * wg + bgv);
    for (int c = 0; c < cpg; ++c) {
      float4 o = q[c];
      o.x *= sg.x; o.y *= sg.y; o.z *= sg.z; o.w *= sg.w;
      ((float4*)(og + (size_t)c * HW))[i] = o;
    }
  }
}

// ---------------------------------------------------------------------------
// Fused 1x1-conv / projection GEMM on WMMA f16 (f32 accumulate); all layout
// logic is compile-time. Block = 128 threads = 4 waves sharing one B tile
// (32K x 16 pixels); each wave owns a 16-row M tile.
//  WTRANS=0: W is (Cout,K) -> A fragment loaded DIRECTLY from global (8x
//            float2, rows are contiguous and L2-hot); no A staging.
//  WTRANS=1: W is (K,Cout) -> A staged via LDS [k][m] stride 20.
//  ACT: 0=none, 1=SiLU, 2=ReLU + residual add.  TAILM: Cout%16 tail guard.
// ---------------------------------------------------------------------------
template<int WTRANS, int INNHWC, int OUTNHWC, int ACT, int SPLIT,
         int HASG, int HASB, int TAILM>
__global__ __launch_bounds__(128) void gemm1x1_wmma(
    const float* __restrict__ Wmat,
    const float* __restrict__ in0, int Cin0,
    const float* __restrict__ in1, int Cin1, int splitK,
    const float* __restrict__ gamma,
    const float* __restrict__ beta,
    const float* __restrict__ resid,
    float* __restrict__ out,
    int Cout, int K, int P, int HW)
{
  __shared__ float ldsB[640];                       // NCHW: [k][pix] s20; NHWC: [pix][k] s36
  __shared__ float ldsA[WTRANS ? 4 * 640 : 4];      // only used when WTRANS=1
  const int tid  = threadIdx.x;
  const int lane = tid & 31;
  const int wv   = tid >> 5;
  const int lh   = lane & 15;
  const int hi   = lane >> 4;
  const int p0   = blockIdx.x * 16;
  const int m0   = (blockIdx.y * 4 + wv) * 16;
  const bool active = (m0 < Cout);

  const int n  = p0 / HW;            // 16-pixel tile never crosses a batch row
  const int s0 = p0 - n * HW;

  float* myA = &ldsA[WTRANS ? wv * 640 : 0];
  const int   mA    = m0 + lh;
  const int   mAc   = min(mA, Cout - 1);
  const float wmask = (mA < Cout) ? 1.f : 0.f;

  v8f acc = {};
  for (int kb = 0; kb < K; kb += 32) {
    // ---- stage B tile cooperatively (all 128 threads, float4 each) ----
    if (INNHWC) {
      const int px   = tid >> 3;          // 0..15
      const int part = (tid & 7) * 4;     // 0..28
      const float4 q = *(const float4*)&in0[(size_t)(p0 + px) * Cin0 + kb + part];
      *(float4*)&ldsB[px * 36 + part] = q;
    } else {
      const int krow = tid >> 2;          // 0..31
      const int part = (tid & 3) * 4;     // 0..12
      const float* src = in0;
      int kk = kb + krow, Cin = Cin0;
      if (SPLIT && kk >= splitK) { src = in1; kk -= splitK; Cin = Cin1; }
      const float4 q = *(const float4*)&src[((size_t)n * Cin + kk) * HW + s0 + part];
      *(float4*)&ldsB[krow * 20 + part] = q;
      if (kb + 32 < K) {                  // prefetch next K slice
        const float* ps = in0;
        int kn = kb + 32 + krow, Cn = Cin0;
        if (SPLIT && kn >= splitK) { ps = in1; kn -= splitK; Cn = Cin1; }
        __builtin_prefetch(&ps[((size_t)n * Cn + kn) * HW + s0 + part], 0, 0);
      }
    }
    // ---- stage A tile in LDS (only transposed-weight layout needs it) ----
    if (WTRANS && active) {
      const int krow = kb + lane;
      if (!TAILM) {
        const float* wr = &Wmat[(size_t)krow * Cout + m0];
#pragma unroll
        for (int q4 = 0; q4 < 4; ++q4)
          *(float4*)&myA[lane * 20 + q4 * 4] = *(const float4*)&wr[q4 * 4];
      } else {                            // tail M-tile (e.g. Cout=72)
        if (m0 + 16 <= Cout) {
          const float* wr = &Wmat[(size_t)krow * Cout + m0];
#pragma unroll
          for (int q4 = 0; q4 < 4; ++q4)
            *(float4*)&myA[lane * 20 + q4 * 4] = *(const float4*)&wr[q4 * 4];
        } else {
#pragma unroll
          for (int e = 0; e < 16; ++e) {
            const int m = m0 + e;
            myA[lane * 20 + e] = (m < Cout) ? Wmat[(size_t)krow * Cout + m] : 0.f;
          }
        }
      }
    }
    __syncthreads();
    // ---- build fragments, issue WMMA ----
    if (active) {
      v16h a, bf;
      if (WTRANS) {                       // A from LDS [k][m]
#pragma unroll
        for (int j = 0; j < 16; ++j) {
          const int ka = (j & 7) + ((j >> 3) << 4) + hi * 8;
          a[j] = (_Float16)myA[ka * 20 + lh];
        }
      } else {                            // A directly from global: 8x float2
        const float* wr = &Wmat[(size_t)mAc * K + kb + hi * 8];
#pragma unroll
        for (int r = 0; r < 8; ++r) {
          const int ko = 2 * (r & 3) + ((r >> 2) << 4);   // 0,2,4,6,16,18,20,22
          const float2 q = *(const float2*)&wr[ko];
          a[2 * r]     = (_Float16)(q.x * wmask);
          a[2 * r + 1] = (_Float16)(q.y * wmask);
        }
      }
      if (INNHWC) {                       // B from LDS [pix][k]: 4x float4
#pragma unroll
        for (int q4 = 0; q4 < 4; ++q4) {
          const float4 q = *(const float4*)&ldsB[lh * 36 + hi * 8 + (q4 & 1) * 4 + (q4 >> 1) * 16];
          bf[q4 * 4 + 0] = (_Float16)q.x; bf[q4 * 4 + 1] = (_Float16)q.y;
          bf[q4 * 4 + 2] = (_Float16)q.z; bf[q4 * 4 + 3] = (_Float16)q.w;
        }
      } else {                            // B from LDS [k][pix]
#pragma unroll
        for (int j = 0; j < 16; ++j)
          bf[j] = (_Float16)ldsB[((j & 7) + ((j >> 3) << 4) + hi * 8) * 20 + lh];
      }
      acc = __builtin_amdgcn_wmma_f32_16x16x32_f16(
          false, a, false, bf, (short)0, acc, false, false);
    }
    __syncthreads();
  }
  if (!active) return;

  // C layout: lane<16 -> rows r, lane>=16 -> rows 8+r; col = lane%16
  const int pcol = p0 + lh;
  const int sc   = s0 + lh;
  if (OUTNHWC) {
    const int mb = m0 + hi * 8;           // 8 consecutive channels
    if (!TAILM || mb + 8 <= Cout) {       // Cout=72 tail aligns with hi split
      float o[8];
#pragma unroll
      for (int r = 0; r < 8; ++r) {
        float v = acc[r];
        if (HASG) v *= gamma[mb + r];
        if (HASB) v += beta[mb + r];
        if (ACT == 1) v *= fast_sigmoid(v);
        o[r] = v;
      }
      float* op = &out[(size_t)pcol * Cout + mb];
      *(float4*)&op[0] = *(float4*)&o[0];
      *(float4*)&op[4] = *(float4*)&o[4];
    }
  } else {
#pragma unroll
    for (int r = 0; r < 8; ++r) {
      const int m = m0 + r + hi * 8;
      if (TAILM && m >= Cout) continue;
      float v = acc[r];
      if (HASG) v *= gamma[m];
      if (HASB) v += beta[m];
      if (ACT == 1) v *= fast_sigmoid(v);                  // SiLU
      const size_t oaddr = ((size_t)n * Cout + m) * HW + sc;
      if (ACT == 2) v = fmaxf(v, 0.f) + resid[oaddr];      // ReLU + residual
      out[oaddr] = v;
    }
  }
}

// ---------------------------------------------------------------------------
// Bilinear 2x upsample (align_corners=False), NCHW.
// ---------------------------------------------------------------------------
__global__ __launch_bounds__(256) void upsample_kernel(
    const float* __restrict__ in, float* __restrict__ out,
    int NC, int IH, int IW, int OH, int OW)
{
  const int idx = blockIdx.x * blockDim.x + threadIdx.x;
  const int total = NC * OH * OW;
  if (idx >= total) return;
  const int ox = idx % OW;
  int t = idx / OW;
  const int oy = t % OH;
  const int nc = t / OH;
  const float ys = fminf(fmaxf((oy + 0.5f) * ((float)IH / OH) - 0.5f, 0.f), IH - 1.f);
  const float xs = fminf(fmaxf((ox + 0.5f) * ((float)IW / OW) - 0.5f, 0.f), IW - 1.f);
  const int y0 = (int)floorf(ys), x0 = (int)floorf(xs);
  const int y1 = min(y0 + 1, IH - 1), x1 = min(x0 + 1, IW - 1);
  const float wy = ys - y0, wx = xs - x0;
  const float* p = in + (size_t)nc * IH * IW;
  const float a = p[y0 * IW + x0], b = p[y0 * IW + x1];
  const float c = p[y1 * IW + x0], d = p[y1 * IW + x1];
  out[idx] = a * (1 - wy) * (1 - wx) + b * (1 - wy) * wx
           + c * wy * (1 - wx)       + d * wy * wx;
}

// ---------------------------------------------------------------------------
// DCNv3 sampling: one lane per (n,y,x,g). xp/off/mk/v are NHWC.
// off channel layout: g*18 + kk*2 + {0:dx,1:dy}; mk: g*9 + kk.
// ---------------------------------------------------------------------------
__global__ __launch_bounds__(256) void dcn_sample_kernel(
    const float* __restrict__ xp,   // NHWC, C=128 (8 groups x 16)
    const float* __restrict__ off,  // NHWC, C=144
    const float* __restrict__ mk,   // NHWC, C=72
    float* __restrict__ v,          // NHWC, C=128
    int N, int H, int W)
{
  const int idx = blockIdx.x * blockDim.x + threadIdx.x;
  const int g = idx & 7;
  int t = idx >> 3;
  const int x = t % W; t /= W;
  const int y = t % H; t /= H;
  const int n = t;
  if (n >= N) return;
  const size_t p = (size_t)(n * H + y) * W + x;
  const float* offp = off + p * 144 + g * 18;
  const float* mkp  = mk  + p * 72  + g * 9;

  float e[9], mx = -3.4e38f;
#pragma unroll
  for (int k = 0; k < 9; ++k) { e[k] = mkp[k]; mx = fmaxf(mx, e[k]); }
  float ssum = 0.f;
#pragma unroll
  for (int k = 0; k < 9; ++k) { e[k] = __expf(e[k] - mx); ssum += e[k]; }
  const float rs = __builtin_amdgcn_rcpf(ssum);

  float4 vo0 = {0,0,0,0}, vo1 = {0,0,0,0}, vo2 = {0,0,0,0}, vo3 = {0,0,0,0};
#pragma unroll
  for (int k = 0; k < 9; ++k) {
    const int dy = k / 3 - 1, dx = k % 3 - 1;
    const float px = (float)x + dx + offp[k * 2 + 0];
    const float py = (float)y + dy + offp[k * 2 + 1];
    const float x0f = floorf(px), y0f = floorf(py);
    const float lx = px - x0f, ly = py - y0f;
    const int x0 = (int)x0f, y0 = (int)y0f;
    const float wgt = e[k] * rs;
    const float cw[4] = {(1 - ly) * (1 - lx), (1 - ly) * lx,
                         ly * (1 - lx),       ly * lx};
#pragma unroll
    for (int cr = 0; cr < 4; ++cr) {
      const int yy = y0 + (cr >> 1);
      const int xx = x0 + (cr & 1);
      if (yy < 0 || yy >= H || xx < 0 || xx >= W) continue;
      const float wv = cw[cr] * wgt;
      const float4* src = (const float4*)(xp + ((size_t)(n * H + yy) * W + xx) * 128 + g * 16);
      const float4 s0 = src[0], s1 = src[1], s2 = src[2], s3 = src[3];
      vo0.x += s0.x * wv; vo0.y += s0.y * wv; vo0.z += s0.z * wv; vo0.w += s0.w * wv;
      vo1.x += s1.x * wv; vo1.y += s1.y * wv; vo1.z += s1.z * wv; vo1.w += s1.w * wv;
      vo2.x += s2.x * wv; vo2.y += s2.y * wv; vo2.z += s2.z * wv; vo2.w += s2.w * wv;
      vo3.x += s3.x * wv; vo3.y += s3.y * wv; vo3.z += s3.z * wv; vo3.w += s3.w * wv;
    }
  }
  float4* dst = (float4*)(v + p * 128 + g * 16);
  dst[0] = vo0; dst[1] = vo1; dst[2] = vo2; dst[3] = vo3;
}

// ---------------------------------------------------------------------------
// Launch: full pipeline. Workspace regions (floats), lifetime-based reuse:
//   sge0 [0, 8388608)                                alive to the end
//   sge1 [8388608, +4194304) / y1 [12582912, +2097152)  -> reused by mk
//   fu   [14680064, +8388608)                        -> reused by v
//   aid  [23068672, +8388608)
//   xp   [31457280, +8388608)
//   off  [39845888, +9437184)        total ~197 MB of d_ws
// ---------------------------------------------------------------------------
extern "C" void kernel_launch(void* const* d_in, const int* in_sizes, int n_in,
                              void* d_out, int out_size, void* d_ws, size_t ws_size,
                              hipStream_t stream)
{
  const float* x0      = (const float*)d_in[0];
  const float* x1      = (const float*)d_in[1];
  const float* sge_w   = (const float*)d_in[2];
  const float* sge_b   = (const float*)d_in[3];
  const float* conv_w  = (const float*)d_in[4];
  const float* conv_g  = (const float*)d_in[5];
  const float* conv_b  = (const float*)d_in[6];
  const float* conv3_w = (const float*)d_in[7];
  const float* conv3_g = (const float*)d_in[8];
  const float* conv3_b = (const float*)d_in[9];
  const float* Wi      = (const float*)d_in[10];
  const float* bi      = (const float*)d_in[11];
  const float* Woff    = (const float*)d_in[12];
  const float* boff    = (const float*)d_in[13];
  const float* Wm      = (const float*)d_in[14];
  const float* bm      = (const float*)d_in[15];
  const float* Wout    = (const float*)d_in[16];
  const float* bout    = (const float*)d_in[17];

  const int B = 4, H = 128, W = 128, H1 = 64, W1 = 64;
  const int HW = H * W, HW1 = H1 * W1;

  float* ws   = (float*)d_ws;
  float* sge0 = ws;
  float* sge1 = ws + 8388608;
  float* y1   = ws + 12582912;
  float* mk   = ws + 8388608;    // reuses sge1+y1 region (needs 4718592)
  float* fu   = ws + 14680064;
  float* vbuf = fu;              // reuses fu after conv3/xp GEMMs
  float* aid  = ws + 23068672;
  float* xp   = ws + 31457280;
  float* offb = ws + 39845888;

  // 1-2) SGE on both inputs
  sge_kernel<<<B * 64, 256, 0, stream>>>(x0, sge_w, sge_b, sge0, 128 / 64, HW);
  sge_kernel<<<B * 64, 256, 0, stream>>>(x1, sge_w, sge_b, sge1, 256 / 64, HW1);

  // 3) conv 1x1 256->128 + BN + SiLU on sge1 (NCHW)
  gemm1x1_wmma<0,0,0,1,0,1,1,0><<<dim3(B * HW1 / 16, 2), 128, 0, stream>>>(
      conv_w, sge1, 256, nullptr, 0, 256,
      conv_g, conv_b, nullptr, y1, 128, 256, B * HW1, HW1);

  // 4) bilinear 2x upsample -> feat_up (NCHW)
  {
    const int total = B * 128 * HW;
    upsample_kernel<<<(total + 255) / 256, 256, 0, stream>>>(y1, fu, B * 128, H1, W1, H, W);
  }
  // 5) conv3 1x1 on concat(feat_up, sge0) + BN + SiLU -> offset_aid (NCHW)
  gemm1x1_wmma<0,0,0,1,1,1,1,0><<<dim3(B * HW / 16, 2), 128, 0, stream>>>(
      conv3_w, fu, 128, sge0, 128, 128,
      conv3_g, conv3_b, nullptr, aid, 128, 256, B * HW, HW);

  // 6) xp = feat_up @ Wi + bi  (NHWC out for the sampler)
  gemm1x1_wmma<1,0,1,0,0,0,1,0><<<dim3(B * HW / 16, 2), 128, 0, stream>>>(
      Wi, fu, 128, nullptr, 0, 128,
      nullptr, bi, nullptr, xp, 128, 128, B * HW, HW);

  // 7) off = aid @ Woff + boff (NHWC, 144ch)
  gemm1x1_wmma<1,0,1,0,0,0,1,0><<<dim3(B * HW / 16, 3), 128, 0, stream>>>(
      Woff, aid, 128, nullptr, 0, 128,
      nullptr, boff, nullptr, offb, 144, 128, B * HW, HW);

  // 8) mk = aid @ Wm + bm (NHWC, 72ch; softmax in sampler)
  gemm1x1_wmma<1,0,1,0,0,0,1,1><<<dim3(B * HW / 16, 2), 128, 0, stream>>>(
      Wm, aid, 128, nullptr, 0, 128,
      nullptr, bm, nullptr, mk, 72, 128, B * HW, HW);

  // 9) DCNv3 bilinear sampling -> v (NHWC)
  {
    const int total = B * HW * 8;
    dcn_sample_kernel<<<total / 256, 256, 0, stream>>>(xp, offb, mk, vbuf, B, H, W);
  }
  // 10) out = relu(v @ Wout + bout) + sge0  (NCHW f32 to d_out)
  gemm1x1_wmma<1,1,0,2,0,0,1,0><<<dim3(B * HW / 16, 2), 128, 0, stream>>>(
      Wout, vbuf, 128, nullptr, 0, 128,
      nullptr, bout, sge0, (float*)d_out, 128, 128, B * HW, HW);
}
